// DiTBlock_67989332295816
// MI455X (gfx1250) — matrix operand order repaired and checked by reading, not compiled, not run on hardware
//
#include <hip/hip_runtime.h>
#include <hip/hip_bf16.h>

// ---- problem constants (fixed by the reference) ----
#define DIM   512
#define HD    256
#define BB    4
#define LL    8192
#define NN    (BB*LL)      // 32768 positions
#define KDIM  512          // inner dim of every big GEMM
#define KT    16           // K-steps of 32

typedef __attribute__((ext_vector_type(16))) __bf16 v16bf;
typedef __attribute__((ext_vector_type(8)))  float  v8f;

__device__ __forceinline__ unsigned short f2bf(float f) {
  __bf16 h = (__bf16)f;
  return __builtin_bit_cast(unsigned short, h);
}
__device__ __forceinline__ float bf2f(unsigned short u) {
  __bf16 h = __builtin_bit_cast(__bf16, u);
  return (float)h;
}

// Fragment-packed bf16 activation layout ("B fragments" of 16x16x32 WMMA):
//   element (k, n) lives at ((n/16 * 16 + k/32)*32 + (n%16) + (k%32>=16?16:0))*16 + k%16
// so each lane of a consuming wave loads its 16 bf16 as 32 contiguous bytes.
__device__ __forceinline__ size_t frag_slot(int n, int ktile, int chalf) {
  return ((((size_t)(n >> 4) * KT + ktile) * 32) + (n & 15) + chalf) * 16;
}

struct U16x16 { unsigned short u[16]; };
__device__ __forceinline__ U16x16 ld32B(const unsigned short* p) {
  U16x16 r;
  const uint4* q = reinterpret_cast<const uint4*>(p);
  *reinterpret_cast<uint4*>(&r.u[0]) = q[0];
  *reinterpret_cast<uint4*>(&r.u[8]) = q[1];
  return r;
}

// ---------------------------------------------------------------------------
// Conditioning: six mp_w-normalized [512,512] x c[4,512] GEMVs.
// mod layout [mat][b][o]: 0=Ws_scale 1=Ws_shift 2=g_seq*Ws_alpha
//                         3=Wc_scale 4=Wc_shift 5=g_chn*Wc_alpha
// ---------------------------------------------------------------------------
__global__ __launch_bounds__(256)
void mod_kernel(const float* __restrict__ Ws_scale, const float* __restrict__ Ws_shift,
                const float* __restrict__ Ws_alpha, const float* __restrict__ Wc_scale,
                const float* __restrict__ Wc_shift, const float* __restrict__ Wc_alpha,
                const float* __restrict__ c, const float* __restrict__ g_seq,
                const float* __restrict__ g_chn, float* __restrict__ mod)
{
  int t = blockIdx.x * 256 + threadIdx.x;          // 6*4*512 = 12288
  int o = t & 511, b = (t >> 9) & 3, m = t >> 11;
  const float* W = (m==0)?Ws_scale:(m==1)?Ws_shift:(m==2)?Ws_alpha:
                   (m==3)?Wc_scale:(m==4)?Wc_shift:Wc_alpha;
  const float* wr = W + (size_t)o * KDIM;
  const float* cb = c + (size_t)b * KDIM;
  float sn = 0.f, dot = 0.f;
  for (int i = 0; i < KDIM; ++i) { float w = wr[i]; sn += w*w; dot += w*cb[i]; }
  float val = dot * rsqrtf((sn + 1e-8f) * (float)KDIM);
  if (m == 2) val *= g_seq[0];
  if (m == 5) val *= g_chn[0];
  mod[t] = val;
}

// ---------------------------------------------------------------------------
// Global weight row table: rows 0..1023 = [Wz_f|Wh_f|Wz_b|Wh_b],
// 1024 W_seq_out, 1536 W_in, 2048 W_pw_h, 2560 W_pw_g, 3072 W_out. (3584 rows)
// ---------------------------------------------------------------------------
__device__ __forceinline__ const float* map_row(int r,
    const float* Wz_f, const float* Wh_f, const float* Wz_b, const float* Wh_b,
    const float* W_seq_out, const float* W_in, const float* W_pw_h,
    const float* W_pw_g, const float* W_out)
{
  if (r < 1024) {
    int m = r >> 8;
    const float* Wm = (m==0)?Wz_f:(m==1)?Wh_f:(m==2)?Wz_b:Wh_b;
    return Wm + (size_t)(r & 255) * KDIM;
  }
  if (r < 1536) return W_seq_out + (size_t)(r - 1024) * KDIM;
  if (r < 2048) return W_in      + (size_t)(r - 1536) * KDIM;
  if (r < 2560) return W_pw_h    + (size_t)(r - 2048) * KDIM;
  if (r < 3072) return W_pw_g    + (size_t)(r - 2560) * KDIM;
  return W_out + (size_t)(r - 3072) * KDIM;
}

__global__ __launch_bounds__(256)
void rownorm_kernel(const float* Wz_f, const float* Wh_f, const float* Wz_b,
                    const float* Wh_b, const float* W_seq_out, const float* W_in,
                    const float* W_pw_h, const float* W_pw_g, const float* W_out,
                    float* __restrict__ rs)
{
  int r = blockIdx.x * 256 + threadIdx.x;          // 0..3583
  const float* row = map_row(r, Wz_f, Wh_f, Wz_b, Wh_b, W_seq_out, W_in,
                             W_pw_h, W_pw_g, W_out);
  float s = 0.f;
  for (int i = 0; i < KDIM; ++i) { float w = row[i]; s += w*w; }
  rs[r] = rsqrtf((s + 1e-8f) * (float)KDIM);       // 1/(||w||*sqrt(fan_in))
}

// Pack normalized weights into WMMA A-fragment layout (16x32 bf16 per tile):
// frag offset = ((g*16 + kt)*32 + lane)*16; lane row M = g*16 + lane%16;
// lanes 0-15: K = kb+{0..15}; lanes 16-31: K = kb+{16..31}.
__global__ __launch_bounds__(256)
void pack_kernel(const float* Wz_f, const float* Wh_f, const float* Wz_b,
                 const float* Wh_b, const float* W_seq_out, const float* W_in,
                 const float* W_pw_h, const float* W_pw_g, const float* W_out,
                 const float* __restrict__ rs, unsigned short* __restrict__ Wp)
{
  int t = blockIdx.x * 256 + threadIdx.x;          // 224 rowtiles * 16 kt * 32 lanes
  int lane = t & 31, kt = (t >> 5) & 15, g = t >> 9;
  int M = g * 16 + (lane & 15);
  const float* row = map_row(M, Wz_f, Wh_f, Wz_b, Wh_b, W_seq_out, W_in,
                             W_pw_h, W_pw_g, W_out);
  float sc = rs[M];
  int kb = kt * 32;
  int k0 = kb + ((lane < 16) ? 0 : 16);
  union { unsigned short u[16]; uint4 q[2]; } ov;
  #pragma unroll
  for (int j = 0; j < 8; ++j) ov.u[j]     = f2bf(row[k0 + j]      * sc);
  #pragma unroll
  for (int j = 0; j < 8; ++j) ov.u[8 + j] = f2bf(row[k0 + 8 + j] * sc);
  size_t off = (((size_t)g * KT + kt) * 32 + lane) * 16;
  uint4* dst = reinterpret_cast<uint4*>(Wp + off);
  dst[0] = ov.q[0]; dst[1] = ov.q[1];
}

// ---------------------------------------------------------------------------
// pixel_norm + scale/shift; emits fragment-packed bf16 activations.
// Each thread owns one column n and emits 16 channels per b128 store pair.
// ---------------------------------------------------------------------------
__global__ __launch_bounds__(256)
void normmod_kernel(const float* __restrict__ xin, const float* __restrict__ scale,
                    const float* __restrict__ shift, unsigned short* __restrict__ Xp)
{
  int n = blockIdx.x * 256 + threadIdx.x;          // n = b*L + l
  int b = n >> 13, l = n & (LL - 1);
  const float* xp = xin + (size_t)b * DIM * LL + l;
  float ss = 0.f;
  for (int d = 0; d < DIM; ++d) { float v = xp[(size_t)d * LL]; ss += v*v; }
  float rsn = rsqrtf(ss * (1.f / (float)DIM) + 1e-4f);
  const float* sc = scale + b * DIM;
  const float* sh = shift + b * DIM;
  for (int db = 0; db < DIM; db += 16) {
    union { unsigned short u[16]; uint4 q[2]; } ov;
    #pragma unroll
    for (int j = 0; j < 16; ++j) {
      int d = db + j;
      float v = xp[(size_t)d * LL] * rsn * sc[d] + sh[d];
      ov.u[j] = f2bf(v);
    }
    uint4* dst = reinterpret_cast<uint4*>(Xp + frag_slot(n, db >> 5, db & 16));
    dst[0] = ov.q[0]; dst[1] = ov.q[1];
  }
}

// ---------------------------------------------------------------------------
// bf16 WMMA GEMM, all operands fragment-packed in global memory (no LDS,
// no barriers). 128 threads / 4 waves; wave tile = 32(out) x 64(pos):
// per K-step: 12 coalesced global_load_b128 feed 8 v_wmma_f32_16x16x32_bf16.
// ---------------------------------------------------------------------------
enum { EPI_F32 = 0, EPI_PACKB = 1, EPI_RESID = 2, EPI_SILU = 3 };

template<int EPI>
__global__ __launch_bounds__(128)
void gemm_kernel(const unsigned short* __restrict__ Wp,   // packed A frags
                 const unsigned short* __restrict__ Xp,   // packed B frags
                 float* __restrict__ Yf,                   // f32 / residual out
                 unsigned short* __restrict__ Yb,          // packed bf16 out
                 const float* __restrict__ gate,           // [B][512] (RESID)
                 const float* __restrict__ base,           // residual base (RESID)
                 const float* __restrict__ Hmul)           // h-buffer (SILU)
{
  const int tid  = threadIdx.x;
  const int wave = tid >> 5;
  const int lane = tid & 31;
  const int nt0  = blockIdx.x * 4;               // four 16-col tiles
  const int o0   = blockIdx.y * 128 + wave * 32; // two 16-row tiles per wave
  const int g0   = o0 >> 4;
  const int nl   = lane & 15;
  const int hi   = lane >> 4;

  v8f acc[2][4] = {};

  for (int kt = 0; kt < KT; ++kt) {
    const unsigned short* wbase = Wp + (((size_t)g0 * KT + kt) * 32 + lane) * 16;
    v16bf a0 = *reinterpret_cast<const v16bf*>(wbase);
    v16bf a1 = *reinterpret_cast<const v16bf*>(wbase + (size_t)KT * 32 * 16);
    if (kt + 2 < KT) {
      __builtin_prefetch(wbase + (size_t)2 * 32 * 16, 0, 3);
      __builtin_prefetch(Xp + (((size_t)nt0 * KT + kt + 2) * 32 + lane) * 16, 0, 3);
    }
    #pragma unroll
    for (int ct = 0; ct < 4; ++ct) {
      const unsigned short* xb =
          Xp + (((size_t)(nt0 + ct) * KT + kt) * 32 + lane) * 16;
      v16bf bfrag = *reinterpret_cast<const v16bf*>(xb);
      acc[0][ct] = __builtin_amdgcn_wmma_f32_16x16x32_bf16(
          false, a0, false, bfrag, (short)0, acc[0][ct], false, false);
      acc[1][ct] = __builtin_amdgcn_wmma_f32_16x16x32_bf16(
          false, a1, false, bfrag, (short)0, acc[1][ct], false, false);
    }
  }

  #pragma unroll
  for (int rt = 0; rt < 2; ++rt) {
    const int o0t = o0 + rt * 16;
    #pragma unroll
    for (int ct = 0; ct < 4; ++ct) {
      const int nt = nt0 + ct;
      const int n  = nt * 16 + nl;
      if constexpr (EPI == EPI_F32) {
        #pragma unroll
        for (int r = 0; r < 8; ++r) {
          int o = o0t + r + hi * 8;              // D layout: VGPR r -> M=r / r+8
          Yf[(size_t)o * NN + n] = acc[rt][ct][r];
        }
      } else if constexpr (EPI == EPI_RESID) {
        int b = n >> 13, l = n & (LL - 1);
        #pragma unroll
        for (int r = 0; r < 8; ++r) {
          int o = o0t + r + hi * 8;
          size_t idx = ((size_t)(b * DIM + o)) * LL + l;
          Yf[idx] = base[idx] + gate[b * DIM + o] * acc[rt][ct][r];
        }
      } else {  // packed bf16 output (optionally fused mp_silu(g)*h)
        union { unsigned short u[8]; uint4 q; } ov;
        #pragma unroll
        for (int r = 0; r < 8; ++r) {
          float v = acc[rt][ct][r];
          if constexpr (EPI == EPI_SILU) {
            int o = o0t + r + hi * 8;
            v = v / (1.f + __expf(-v)) * (1.0f / 0.596f)
                * Hmul[(size_t)o * NN + n];
          }
          ov.u[r] = f2bf(v);
        }
        // D rows o0t..o0t+15 land exactly in one consumer fragment slot:
        // lane' = nl + (o0t&16), j = hi*8 + r  -> one b128 store per tile.
        size_t addr = ((((size_t)nt * KT + (o0t >> 5)) * 32)
                       + nl + (o0t & 16)) * 16 + hi * 8;
        *reinterpret_cast<uint4*>(Yb + addr) = ov.q;
      }
    }
  }
}

// ---------------------------------------------------------------------------
// MinGRU scan. P[1024][N] f32: rows dir*512+j = z-pre, dir*512+256+j = h-pre.
// One block per (dir,b,j): chunked serial compose + LDS compose-scan.
// h_t = (1-z)h_{t-1} + z*h~ ; compose (A,B)∘(a,b) = (a*A, a*B + b).
// Output Hcat fragment-packed (row = K index of the W_seq_out GEMM).
// ---------------------------------------------------------------------------
__global__ __launch_bounds__(256)
void scan_kernel(const float* __restrict__ P, unsigned short* __restrict__ Hcat)
{
  int blk = blockIdx.x;                        // 2048 = dir*1024 + b*256 + j
  int j = blk & 255, b = (blk >> 8) & 3, dir = blk >> 10;
  const float* Pz = P + (size_t)(dir * 512 + j)       * NN + b * LL;
  const float* Ph = P + (size_t)(dir * 512 + 256 + j) * NN + b * LL;
  const int row   = dir * 256 + j;             // K index in Hcat
  const int ktr   = row >> 5, chalf = row & 16, jj = row & 15;

  int t = threadIdx.x;
  int s0 = t * 32;
  float A = 1.f, Bv = 0.f;
  for (int i = 0; i < 32; ++i) {
    int s = s0 + i; int l = dir ? (LL - 1 - s) : s;
    float z = 1.f / (1.f + __expf(-Pz[l]));
    float a = 1.f - z;
    A = a * A; Bv = a * Bv + z * Ph[l];
  }
  __shared__ float sA[256], sB[256];
  sA[t] = A; sB[t] = Bv;
  __syncthreads();
  for (int off = 1; off < 256; off <<= 1) {    // Hillis-Steele inclusive scan
    float pa = 1.f, pb = 0.f;
    if (t >= off) { pa = sA[t - off]; pb = sB[t - off]; }
    __syncthreads();
    float ca = sA[t], cb = sB[t];
    sA[t] = pa * ca; sB[t] = pb * ca + cb;
    __syncthreads();
  }
  float h = (t == 0) ? 0.f : sB[t - 1];        // exclusive prefix, h0 = 0
  for (int i = 0; i < 32; ++i) {
    int s = s0 + i; int l = dir ? (LL - 1 - s) : s;
    float z = 1.f / (1.f + __expf(-Pz[l]));
    h = (1.f - z) * h + z * Ph[l];
    int n = b * LL + l;
    Hcat[frag_slot(n, ktr, chalf) + jj] = f2bf(h);
  }
}

// ---------------------------------------------------------------------------
// Both depthwise k=3 convs (mp_w-normalized taps), zero pad per-b segment.
// Fragment-packed in/out: each thread owns one 16-channel slot of one column,
// 3x b128-pair loads -> 2x b128-pair stores. Taps staged in LDS.
// ---------------------------------------------------------------------------
__global__ __launch_bounds__(512)
void dwconv_kernel(const unsigned short* __restrict__ R1,
                   const float* __restrict__ Wdh, const float* __restrict__ Wdg,
                   unsigned short* __restrict__ Dh, unsigned short* __restrict__ Dg)
{
  __shared__ float th0[512], th1[512], th2[512];
  __shared__ float tg0[512], tg1[512], tg2[512];
  {
    int cc = threadIdx.x;                      // 512 threads, one channel each
    float a0 = Wdh[cc*3], a1 = Wdh[cc*3+1], a2 = Wdh[cc*3+2];
    float b0 = Wdg[cc*3], b1 = Wdg[cc*3+1], b2 = Wdg[cc*3+2];
    float ih = rsqrtf((a0*a0 + a1*a1 + a2*a2 + 1e-8f) * 3.f);
    float ig = rsqrtf((b0*b0 + b1*b1 + b2*b2 + 1e-8f) * 3.f);
    th0[cc] = a0*ih; th1[cc] = a1*ih; th2[cc] = a2*ih;
    tg0[cc] = b0*ig; tg1[cc] = b1*ig; tg2[cc] = b2*ig;
  }
  __syncthreads();

  int t = threadIdx.x;
  int ktp = t >> 5, lanep = t & 31;
  int nl = lanep & 15, chalf = lanep & 16;
  int nt = blockIdx.x;                         // 2048 column tiles
  int n = nt * 16 + nl;
  int l = n & (LL - 1);
  int cbase = ktp * 32 + chalf;

  U16x16 c0 = ld32B(R1 + frag_slot(n, ktp, chalf));
  U16x16 cm = {}, cp = {};
  if (l > 0)      cm = ld32B(R1 + frag_slot(n - 1, ktp, chalf));
  if (l < LL - 1) cp = ld32B(R1 + frag_slot(n + 1, ktp, chalf));

  union { unsigned short u[16]; uint4 q[2]; } oh, og;
  #pragma unroll
  for (int jv = 0; jv < 16; ++jv) {
    int cc = cbase + jv;
    float xm = bf2f(cm.u[jv]), x0 = bf2f(c0.u[jv]), xq = bf2f(cp.u[jv]);
    oh.u[jv] = f2bf(th0[cc]*xm + th1[cc]*x0 + th2[cc]*xq);
    og.u[jv] = f2bf(tg0[cc]*xm + tg1[cc]*x0 + tg2[cc]*xq);
  }
  uint4* dh = reinterpret_cast<uint4*>(Dh + frag_slot(n, ktp, chalf));
  dh[0] = oh.q[0]; dh[1] = oh.q[1];
  uint4* dg = reinterpret_cast<uint4*>(Dg + frag_slot(n, ktp, chalf));
  dg[0] = og.q[0]; dg[1] = og.q[1];
}

// ---------------------------------------------------------------------------
extern "C" void kernel_launch(void* const* d_in, const int* in_sizes, int n_in,
                              void* d_out, int out_size, void* d_ws, size_t ws_size,
                              hipStream_t stream)
{
  const float* x        = (const float*)d_in[0];
  const float* c        = (const float*)d_in[1];
  const float* Ws_scale = (const float*)d_in[2];
  const float* Ws_shift = (const float*)d_in[3];
  const float* Ws_alpha = (const float*)d_in[4];
  const float* g_seq    = (const float*)d_in[5];
  const float* Wz_f     = (const float*)d_in[6];
  const float* Wh_f     = (const float*)d_in[7];
  const float* Wz_b     = (const float*)d_in[8];
  const float* Wh_b     = (const float*)d_in[9];
  const float* W_seqo   = (const float*)d_in[10];
  const float* Wc_scale = (const float*)d_in[11];
  const float* Wc_shift = (const float*)d_in[12];
  const float* Wc_alpha = (const float*)d_in[13];
  const float* g_chn    = (const float*)d_in[14];
  const float* W_in     = (const float*)d_in[15];
  const float* W_dw_h   = (const float*)d_in[16];
  const float* W_pw_h   = (const float*)d_in[17];
  const float* W_dw_g   = (const float*)d_in[18];
  const float* W_pw_g   = (const float*)d_in[19];
  const float* W_out    = (const float*)d_in[20];
  float* xout = (float*)d_out;

  char* ws = (char*)d_ws;
  const size_t MB = 1024ull * 1024ull;
  unsigned short* Wp   = (unsigned short*)(ws + 0);              // 3.5 MB packed weights
  float*          rown = (float*)         (ws + 4*MB);           // 14 KB
  float*          mod  = (float*)         (ws + 4*MB + 64*1024); // 48 KB
  unsigned short* Xp   = (unsigned short*)(ws + 8*MB);           // 32 MB packed acts
  float*          P    = (float*)         (ws + 40*MB);          // 128 MB z/h-pre
  unsigned short* Hcat = (unsigned short*)(ws + 168*MB);         // 32 MB packed
  float*          x1   = (float*)         (ws + 200*MB);         // 64 MB updated x
  // stream-ordered aliases (P free after scan, Hcat free after seq GEMM,
  // Xp free after the W_in GEMM)
  unsigned short* R1   = (unsigned short*)(ws + 40*MB);          // packed
  unsigned short* Dh   = (unsigned short*)(ws + 72*MB);          // packed
  unsigned short* Dg   = (unsigned short*)(ws + 104*MB);         // packed
  float*          Hbuf = (float*)         (ws + 136*MB);         // 64 MB plain f32
  unsigned short* HG   = (unsigned short*)(ws + 8*MB);           // packed

  float* mscale_s = mod + 0;
  float* mshift_s = mod + 2048;
  float* mgate_s  = mod + 4096;
  float* mscale_c = mod + 6144;
  float* mshift_c = mod + 8192;
  float* mgate_c  = mod + 10240;

  // conditioning + weight prep
  mod_kernel<<<48, 256, 0, stream>>>(Ws_scale, Ws_shift, Ws_alpha,
                                     Wc_scale, Wc_shift, Wc_alpha,
                                     c, g_seq, g_chn, mod);
  rownorm_kernel<<<14, 256, 0, stream>>>(Wz_f, Wh_f, Wz_b, Wh_b, W_seqo, W_in,
                                         W_pw_h, W_pw_g, W_out, rown);
  pack_kernel<<<448, 256, 0, stream>>>(Wz_f, Wh_f, Wz_b, Wh_b, W_seqo, W_in,
                                       W_pw_h, W_pw_g, W_out, rown, Wp);

  // ---- sequence-mixer block ----
  normmod_kernel<<<NN/256, 256, 0, stream>>>(x, mscale_s, mshift_s, Xp);
  gemm_kernel<EPI_F32><<<dim3(NN/64, 8), 128, 0, stream>>>(      // O=1024
      Wp + (size_t)0 * KDIM, Xp, P, nullptr, nullptr, nullptr, nullptr);
  scan_kernel<<<2048, 256, 0, stream>>>(P, Hcat);
  gemm_kernel<EPI_RESID><<<dim3(NN/64, 4), 128, 0, stream>>>(    // O=512
      Wp + (size_t)1024 * KDIM, Hcat, x1, nullptr, mgate_s, x, nullptr);

  // ---- channel-mixer block ----
  normmod_kernel<<<NN/256, 256, 0, stream>>>(x1, mscale_c, mshift_c, Xp);
  gemm_kernel<EPI_PACKB><<<dim3(NN/64, 4), 128, 0, stream>>>(
      Wp + (size_t)1536 * KDIM, Xp, nullptr, R1, nullptr, nullptr, nullptr);
  dwconv_kernel<<<2048, 512, 0, stream>>>(R1, W_dw_h, W_dw_g, Dh, Dg);
  gemm_kernel<EPI_F32><<<dim3(NN/64, 4), 128, 0, stream>>>(
      Wp + (size_t)2048 * KDIM, Dh, Hbuf, nullptr, nullptr, nullptr, nullptr);
  gemm_kernel<EPI_SILU><<<dim3(NN/64, 4), 128, 0, stream>>>(
      Wp + (size_t)2560 * KDIM, Dg, nullptr, HG, nullptr, nullptr, Hbuf);
  gemm_kernel<EPI_RESID><<<dim3(NN/64, 4), 128, 0, stream>>>(
      Wp + (size_t)3072 * KDIM, HG, xout, nullptr, mgate_c, x1, nullptr);
}
